// VectorQuantizer_66477503807875
// MI455X (gfx1250) — matrix operand order repaired and checked by reading, not compiled, not run on hardware
//
#include <hip/hip_runtime.h>

// ---------------- problem constants ----------------
#define NREC      65536        // B*D*H*W
#define DM        256          // embedding dim
#define KCB       1024         // codebook size
#define CH_STRIDE 16384        // D*H*W
#define B_STRIDE  4194304      // C*D*H*W
#define ZSZ       16777216     // total elements of z
#define LOSS_OFF  4194304      // d_out layout: [z_q | loss | indices]
#define IDX_OFF   4194305
#define DECAY_F   0.99f
#define EPS_F     1e-5f

typedef __attribute__((ext_vector_type(16))) __bf16  v16bf;
typedef __attribute__((ext_vector_type(8)))  float   v8f;
typedef __attribute__((ext_vector_type(4)))  float   f32x4;
typedef __attribute__((ext_vector_type(4)))  unsigned int u32x4;

union BFrag {
    v16bf v;
    unsigned short u[16];
    u32x4 q[2];
};

__device__ __forceinline__ unsigned short f2bf(float f) {
    unsigned int u = __float_as_uint(f);
    unsigned int r = u + 0x7fffu + ((u >> 16) & 1u);   // round-to-nearest-even
    return (unsigned short)(r >> 16);
}

// ---------------- kernel 1: prep -------------------
// bf16 codebook copy, ||e||^2, zero accumulators/loss.
__global__ void vq_prep(const float* __restrict__ embed,
                        unsigned short* __restrict__ embB,
                        float* __restrict__ eNorm,
                        float* __restrict__ bcs,
                        float* __restrict__ bes,
                        float* __restrict__ nsum,
                        float* __restrict__ outF) {
    int j = blockIdx.x, c = threadIdx.x;
    float e = embed[j * DM + c];
    embB[j * DM + c] = f2bf(e);
    bes[j * DM + c]  = 0.0f;
    __shared__ float red[256];
    red[c] = e * e;
    __syncthreads();
    for (int s = 128; s > 0; s >>= 1) {
        if (c < s) red[c] += red[c + s];
        __syncthreads();
    }
    if (c == 0) {
        eNorm[j] = red[0];
        bcs[j]   = 0.0f;
        if (j == 0) { nsum[0] = 0.0f; outF[LOSS_OFF] = 0.0f; }
    }
}

// ---------------- kernel 2: distances + argmin + EMA scatter -----
// One block = 16 rows of zf. A-tile staged in LDS *pre-swizzled into WMMA
// fragment order* so each lane's per-kk fragment is 32 contiguous bytes
// (2x ds_load_b128); all 8 fragments hoisted into registers before the
// column loop. Inner loop = 2x global_load_b128 (B) + 1x v_wmma.
__global__ void vq_dist(const float* __restrict__ z,
                        const unsigned short* __restrict__ embB,
                        const float* __restrict__ eNorm,
                        int* __restrict__ idxWs,
                        float* __restrict__ bcs,
                        float* __restrict__ bes,
                        float* __restrict__ outF) {
    __shared__ unsigned short lAf[8 * 32 * 16]; // [kk][lane][16] fragment order, 8KB
    __shared__ float sc[16 * 128];              // scores for current column block
    __shared__ float candS[16 * 16];
    __shared__ int   candI[16 * 16];
    __shared__ int   sIdx[16];                  // final argmin per row

    const int t    = threadIdx.x;               // 0..255
    const int n0   = blockIdx.x * 16;           // first row of tile
    const int b    = n0 >> 14;
    const int spat = n0 & 16383;

    // ---- stage A tile: thread t = channel c; 16 contiguous rows (coalesced).
    // Keep the fp32 values in registers for the EMA scatter at the end.
    float zv[16];
    {
        const float* src = z + b * B_STRIDE + t * CH_STRIDE + spat;
        const f32x4* s4 = (const f32x4*)src;
        f32x4 f0 = s4[0], f1 = s4[1], f2 = s4[2], f3 = s4[3];
        zv[0] = f0.x;  zv[1] = f0.y;  zv[2]  = f0.z;  zv[3]  = f0.w;
        zv[4] = f1.x;  zv[5] = f1.y;  zv[6]  = f1.z;  zv[7]  = f1.w;
        zv[8] = f2.x;  zv[9] = f2.y;  zv[10] = f2.z;  zv[11] = f2.w;
        zv[12] = f3.x; zv[13] = f3.y; zv[14] = f3.z;  zv[15] = f3.w;
        // fragment-order scatter: k = t. kr bits: [4]=v2, [3]=half, [2:1]=v10, [0]=p
        const int kk    = t >> 5;
        const int kr    = t & 31;
        const int halfc = (kr >> 3) & 1;
        const int v     = ((kr & 16) >> 2) | ((kr >> 1) & 3);
        const int i     = 2 * v + (kr & 1);
        unsigned short* dst = &lAf[kk * 512 + halfc * 256 + i];
        #pragma unroll
        for (int m = 0; m < 16; ++m) dst[m * 16] = f2bf(zv[m]);
    }
    __syncthreads();

    const int lane = t & 31, wave = t >> 5;
    const int m = lane & 15, half = lane >> 4;

    // ---- hoist all 8 A fragments into registers (2x ds_load_b128 each) ----
    BFrag aF[8];
    {
        const u32x4* lp = (const u32x4*)&lAf[0];
        #pragma unroll
        for (int kk = 0; kk < 8; ++kk) {
            int off = (kk * 32 + lane) * 2;      // in 16B units
            aF[kk].q[0] = lp[off];
            aF[kk].q[1] = lp[off + 1];
        }
    }

    float bs = 3.4e38f;
    int   bi = 0;

    for (int cb = 0; cb < 8; ++cb) {
        const int col = cb * 128 + wave * 16 + m;   // codebook row for B-column
        v8f acc = {};
        #pragma unroll
        for (int kk = 0; kk < 8; ++kk) {
            // B fragment: lane holds column 'col', K = 8*half+0..7 and 16+8*half+0..7 (L2-hot)
            BFrag bf;
            const u32x4* bp = (const u32x4*)(embB + col * DM + kk * 32 + half * 8);
            bf.q[0] = bp[0];
            bf.q[1] = bp[2];
            acc = __builtin_amdgcn_wmma_f32_16x16x32_bf16(
                false, aF[kk].v, false, bf.v, (short)0, acc, false, false);
        }
        // score = ||e||^2 - 2 z.e  (||z||^2 row-constant, dropped for argmin)
        float en = eNorm[col];
        #pragma unroll
        for (int g = 0; g < 8; ++g) {
            int row = g + (half << 3);
            sc[row * 128 + wave * 16 + m] = en - 2.0f * acc[g];
        }
        __syncthreads();
        // parallel argmin: 16 threads/row produce 16 candidates, then reduce
        {
            const int r = t >> 4, cg = t & 15;
            float ls = 3.4e38f; int li = 0;
            #pragma unroll
            for (int q = 0; q < 8; ++q) {
                int c2 = cg * 8 + q;
                float s = sc[r * 128 + c2];
                if (s < ls) { ls = s; li = c2; }
            }
            candS[r * 16 + cg] = ls;
            candI[r * 16 + cg] = cb * 128 + li;
        }
        __syncthreads();
        if (t < 16) {
            #pragma unroll
            for (int c2 = 0; c2 < 16; ++c2) {
                float s = candS[t * 16 + c2];
                if (s < bs) { bs = s; bi = candI[t * 16 + c2]; }  // first-min ties
            }
        }
        __syncthreads();
    }
    if (t < 16) {
        idxWs[n0 + t] = bi;
        outF[IDX_OFF + n0 + t] = (float)bi;
        sIdx[t] = bi;
        atomicAdd(&bcs[bi], 1.0f);
    }
    __syncthreads();
    // ---- fused EMA embed-sum scatter: thread t = channel, rows 0..15 ----
    #pragma unroll
    for (int mm = 0; mm < 16; ++mm) {
        atomicAdd(&bes[sIdx[mm] * DM + t], zv[mm]);
    }
}

// ---------------- kernel 3: EMA update -------------
__global__ void vq_ema(const float* __restrict__ ema_cs,
                       const float* __restrict__ ema_es,
                       const float* __restrict__ bcs,
                       const float* __restrict__ bes,
                       float* __restrict__ newCS,
                       float* __restrict__ newES,
                       float* __restrict__ nsum) {
    int j = blockIdx.x, c = threadIdx.x;
    newES[j * DM + c] = DECAY_F * ema_es[j * DM + c] + (1.0f - DECAY_F) * bes[j * DM + c];
    if (c == 0) {
        float ncs = DECAY_F * ema_cs[j] + (1.0f - DECAY_F) * bcs[j];
        newCS[j] = ncs;
        atomicAdd(nsum, ncs);
    }
}

// ---------------- kernel 4: normalized codebook ----
__global__ void vq_embed(const float* __restrict__ newCS,
                         const float* __restrict__ newES,
                         const float* __restrict__ nsum,
                         float* __restrict__ newEmbed) {
    int j = blockIdx.x, c = threadIdx.x;
    float n = nsum[0];
    float csz = (newCS[j] + EPS_F) / (n + (float)KCB * EPS_F) * n;
    newEmbed[j * DM + c] = newES[j * DM + c] / csz;
}

// ---------------- kernel 5: gather + loss ----------
__global__ void vq_out(const float* __restrict__ z,
                       const int* __restrict__ idxWs,
                       const float* __restrict__ newEmbed,
                       float* __restrict__ outF) {
    const int n0 = blockIdx.x * 16;
    const int r  = threadIdx.x & 15, cg = threadIdx.x >> 4;
    const int j  = idxWs[n0 + r];
    const int base = (n0 >> 14) * B_STRIDE + (n0 & 16383) + r;
    float local = 0.0f;
    #pragma unroll 4
    for (int c = cg * 16; c < cg * 16 + 16; ++c) {
        float e  = newEmbed[j * DM + c];
        int pos  = base + c * CH_STRIDE;
        float zz = z[pos];
        outF[pos] = e;                 // straight-through: z_q numerically
        float d = zz - e;
        local += d * d;
    }
    __shared__ float red[256];
    red[threadIdx.x] = local;
    __syncthreads();
    for (int s = 128; s > 0; s >>= 1) {
        if (threadIdx.x < s) red[threadIdx.x] += red[threadIdx.x + s];
        __syncthreads();
    }
    if (threadIdx.x == 0)
        atomicAdd(&outF[LOSS_OFF], red[0] * (0.25f / (float)ZSZ));
}

// ---------------- launcher -------------------------
extern "C" void kernel_launch(void* const* d_in, const int* in_sizes, int n_in,
                              void* d_out, int out_size, void* d_ws, size_t ws_size,
                              hipStream_t stream) {
    const float* z       = (const float*)d_in[0];
    const float* embed   = (const float*)d_in[1];
    const float* ema_cs  = (const float*)d_in[2];
    const float* ema_es  = (const float*)d_in[3];
    float* outF = (float*)d_out;

    char* ws = (char*)d_ws;
    unsigned short* embB  = (unsigned short*)(ws);              // 512 KB
    float* eNorm          = (float*)(ws + 524288);              // 4 KB
    int*   idxWs          = (int*)  (ws + 528384);              // 256 KB
    float* bcs            = (float*)(ws + 790528);              // 4 KB
    float* bes            = (float*)(ws + 794624);              // 1 MB
    float* newCS          = (float*)(ws + 1843200);             // 4 KB
    float* newES          = (float*)(ws + 1847296);             // 1 MB
    float* newEmbed       = (float*)(ws + 2895872);             // 1 MB
    float* nsum           = (float*)(ws + 3944448);             // 4 B

    (void)in_sizes; (void)n_in; (void)out_size; (void)ws_size;

    vq_prep <<<KCB,       256, 0, stream>>>(embed, embB, eNorm, bcs, bes, nsum, outF);
    vq_dist <<<NREC / 16, 256, 0, stream>>>(z, embB, eNorm, idxWs, bcs, bes, outF);
    vq_ema  <<<KCB,       256, 0, stream>>>(ema_cs, ema_es, bcs, bes, newCS, newES, nsum);
    vq_embed<<<KCB,       256, 0, stream>>>(newCS, newES, nsum, newEmbed);
    vq_out  <<<NREC / 16, 256, 0, stream>>>(z, idxWs, newEmbed, outF);
}
